// KDABlock_50929722196117
// MI455X (gfx1250) — matrix operand order repaired and checked by reading, not compiled, not run on hardware
//
#include <hip/hip_runtime.h>

// Problem constants (match reference)
#define B_  2
#define T_  2048
#define D_  1024
#define H_  16
#define DK_ 64
#define DV_ 64
#define K_  1024
#define V_  1024
#define I_  2816
#define M_  (B_ * T_)   // 4096 rows

typedef __attribute__((ext_vector_type(16))) __bf16 v16bf;
typedef __attribute__((ext_vector_type(8)))  float  v8f;

union Frag {
  v16bf v;
  uint4 q[2];
  unsigned short u[16];
};
union Chunk8 {
  uint4 q;
  unsigned short h[8];
};

__device__ __forceinline__ unsigned short f2bf(float f) {
  unsigned int u = __float_as_uint(f);
  u += 0x7FFFu + ((u >> 16) & 1u);   // round-to-nearest-even
  return (unsigned short)(u >> 16);
}
__device__ __forceinline__ float sigmoidf_(float x) { return 1.f / (1.f + __expf(-x)); }

// ---------------------------------------------------------------- convert
__global__ void convert_f32_bf16(const float* __restrict__ in,
                                 unsigned short* __restrict__ out, int n) {
  int i = blockIdx.x * blockDim.x + threadIdx.x;
  int stride = gridDim.x * blockDim.x;
  for (; i < n; i += stride) out[i] = f2bf(in[i]);
}

// ---------------------------------------------------------------- layernorm -> bf16
__global__ __launch_bounds__(256) void layernorm_kernel(
    const float* __restrict__ x, const float* __restrict__ w,
    const float* __restrict__ b, unsigned short* __restrict__ out) {
  int row = blockIdx.x, tid = threadIdx.x;
  const float* xr = x + (size_t)row * D_;
  float v4[4]; float s = 0.f, sq = 0.f;
#pragma unroll
  for (int i = 0; i < 4; i++) { float t = xr[tid + i * 256]; v4[i] = t; s += t; sq += t * t; }
  __shared__ float ss[256], s2[256];
  ss[tid] = s; s2[tid] = sq; __syncthreads();
  for (int o = 128; o > 0; o >>= 1) {
    if (tid < o) { ss[tid] += ss[tid + o]; s2[tid] += s2[tid + o]; }
    __syncthreads();
  }
  float mean = ss[0] * (1.f / D_);
  float var  = s2[0] * (1.f / D_) - mean * mean;
  float rstd = rsqrtf(var + 1e-5f);
  unsigned short* orow = out + (size_t)row * D_;
#pragma unroll
  for (int i = 0; i < 4; i++) {
    int c = tid + i * 256;
    orow[c] = f2bf((v4[i] - mean) * rstd * w[c] + b[c]);
  }
}

// ---------------------------------------------------------------- generic bf16 WMMA GEMM
// C[M,N] = A[M,Kd](bf16) * Bw[Kd,N](bf16) (+resid), fp32 out.
// 256 thr = 8 waves; 64x64 block tile; wave = two 16x16 WMMA tiles; K-step 32.
// A staged row-major in LDS (fragments = 2x ds_load_b128).
// B staged COLUMN-major in LDS (fragments = 2x ds_load_b128, transpose paid once at staging).
__global__ __launch_bounds__(256) void gemm_bf16_wmma(
    const unsigned short* __restrict__ A, const unsigned short* __restrict__ Bw,
    float* __restrict__ C, const float* __restrict__ resid,
    int M, int N, int Kd) {
  __shared__ __align__(16) unsigned short sA[64][32];    // row-major, 64B row stride
  __shared__ __align__(16) unsigned short sBt[64][40];   // [col][k], 80B row stride
  int tid = threadIdx.x;
  int lane = tid & 31, wid = tid >> 5;
  int rowBase = blockIdx.y * 64, colBase = blockIdx.x * 64;
  int tm = wid >> 1;
  int tn0 = (wid * 2) & 3, tn1 = tn0 + 1;
  v8f c0 = {}, c1 = {};
  int ar = tid >> 2, ac = (tid & 3) * 8;   // A tile: 64x32, one 16B chunk/thread
  int br = tid >> 3, bc = (tid & 7) * 8;   // B tile: 32x64, one 16B chunk/thread
  for (int k0 = 0; k0 < Kd; k0 += 32) {
    *(uint4*)&sA[ar][ac] = *(const uint4*)&A[(size_t)(rowBase + ar) * Kd + k0 + ac];
    int gcol = colBase + bc;
    Chunk8 bz; bz.q = make_uint4(0, 0, 0, 0);
    if (gcol < N) bz.q = *(const uint4*)&Bw[(size_t)(k0 + br) * N + gcol];
#pragma unroll
    for (int j = 0; j < 8; j++) sBt[bc + j][br] = bz.h[j];   // transpose scatter
    if (k0 + 32 < Kd) {
      __builtin_prefetch(&A[(size_t)(rowBase + ar) * Kd + k0 + 32 + ac], 0, 3);
      if (gcol < N) __builtin_prefetch(&Bw[(size_t)(k0 + 32 + br) * N + gcol], 0, 3);
    }
    __syncthreads();
    Frag a, b0, b1;
    int arow = tm * 16 + (lane & 15);
    int kb = (lane >> 4) * 8;            // A: K {0..7,16..23} / {8..15,24..31}
    a.q[0] = *(const uint4*)&sA[arow][kb];
    a.q[1] = *(const uint4*)&sA[arow][16 + kb];
    int kh = (lane >> 4) * 8;            // B: K {0..7,16..23} / {8..15,24..31}
    int c0col = tn0 * 16 + (lane & 15), c1col = tn1 * 16 + (lane & 15);
    b0.q[0] = *(const uint4*)&sBt[c0col][kh];
    b0.q[1] = *(const uint4*)&sBt[c0col][16 + kh];
    b1.q[0] = *(const uint4*)&sBt[c1col][kh];
    b1.q[1] = *(const uint4*)&sBt[c1col][16 + kh];
    c0 = __builtin_amdgcn_wmma_f32_16x16x32_bf16(false, a.v, false, b0.v, (short)0, c0, false, false);
    c1 = __builtin_amdgcn_wmma_f32_16x16x32_bf16(false, a.v, false, b1.v, (short)0, c1, false, false);
    __syncthreads();
  }
  int rb = (lane >> 4) * 8;
  int n0 = colBase + tn0 * 16 + (lane & 15);
  int n1 = colBase + tn1 * 16 + (lane & 15);
#pragma unroll
  for (int r = 0; r < 8; r++) {
    int row = rowBase + tm * 16 + rb + r;
    if (n0 < N) { float v = c0[r]; if (resid) v += resid[(size_t)row * N + n0]; C[(size_t)row * N + n0] = v; }
    if (n1 < N) { float v = c1[r]; if (resid) v += resid[(size_t)row * N + n1]; C[(size_t)row * N + n1] = v; }
  }
}

// ---------------------------------------------------------------- fused gate_up + SwiGLU GEMM
// act[M,I] = silu(A*Wg) * (A*Wu), Bw = [Wg | Wu], ldb = 2I. Output bf16.
__global__ __launch_bounds__(256) void gemm_gateup_wmma(
    const unsigned short* __restrict__ A, const unsigned short* __restrict__ Bw,
    unsigned short* __restrict__ act, int Icols, int Kd) {
  __shared__ __align__(16) unsigned short sA[64][32];
  __shared__ __align__(16) unsigned short sBg[64][40];   // [col][k]
  __shared__ __align__(16) unsigned short sBu[64][40];   // [col][k]
  int tid = threadIdx.x;
  int lane = tid & 31, wid = tid >> 5;
  int rowBase = blockIdx.y * 64, colBase = blockIdx.x * 64;
  int tm = wid >> 1;
  int tn0 = (wid * 2) & 3, tn1 = tn0 + 1;
  v8f cg0 = {}, cg1 = {}, cu0 = {}, cu1 = {};
  int ar = tid >> 2, ac = (tid & 3) * 8;
  int br = tid >> 3, bc = (tid & 7) * 8;
  int ldb = 2 * Icols;
  for (int k0 = 0; k0 < Kd; k0 += 32) {
    *(uint4*)&sA[ar][ac] = *(const uint4*)&A[(size_t)(rowBase + ar) * Kd + k0 + ac];
    size_t bro = (size_t)(k0 + br) * ldb;
    Chunk8 cg, cu;
    cg.q = *(const uint4*)&Bw[bro + colBase + bc];
    cu.q = *(const uint4*)&Bw[bro + Icols + colBase + bc];
#pragma unroll
    for (int j = 0; j < 8; j++) { sBg[bc + j][br] = cg.h[j]; sBu[bc + j][br] = cu.h[j]; }
    __syncthreads();
    Frag a, bg0, bg1, bu0, bu1;
    int arow = tm * 16 + (lane & 15);
    int kb = (lane >> 4) * 8;
    a.q[0] = *(const uint4*)&sA[arow][kb];
    a.q[1] = *(const uint4*)&sA[arow][16 + kb];
    int kh = (lane >> 4) * 8;
    int c0col = tn0 * 16 + (lane & 15), c1col = tn1 * 16 + (lane & 15);
    bg0.q[0] = *(const uint4*)&sBg[c0col][kh]; bg0.q[1] = *(const uint4*)&sBg[c0col][16 + kh];
    bg1.q[0] = *(const uint4*)&sBg[c1col][kh]; bg1.q[1] = *(const uint4*)&sBg[c1col][16 + kh];
    bu0.q[0] = *(const uint4*)&sBu[c0col][kh]; bu0.q[1] = *(const uint4*)&sBu[c0col][16 + kh];
    bu1.q[0] = *(const uint4*)&sBu[c1col][kh]; bu1.q[1] = *(const uint4*)&sBu[c1col][16 + kh];
    cg0 = __builtin_amdgcn_wmma_f32_16x16x32_bf16(false, a.v, false, bg0.v, (short)0, cg0, false, false);
    cg1 = __builtin_amdgcn_wmma_f32_16x16x32_bf16(false, a.v, false, bg1.v, (short)0, cg1, false, false);
    cu0 = __builtin_amdgcn_wmma_f32_16x16x32_bf16(false, a.v, false, bu0.v, (short)0, cu0, false, false);
    cu1 = __builtin_amdgcn_wmma_f32_16x16x32_bf16(false, a.v, false, bu1.v, (short)0, cu1, false, false);
    __syncthreads();
  }
  int rb = (lane >> 4) * 8;
  int n0 = colBase + tn0 * 16 + (lane & 15);
  int n1 = colBase + tn1 * 16 + (lane & 15);
#pragma unroll
  for (int r = 0; r < 8; r++) {
    int row = rowBase + tm * 16 + rb + r;
    float g0 = cg0[r], g1 = cg1[r];
    act[(size_t)row * Icols + n0] = f2bf(g0 * sigmoidf_(g0) * cu0[r]);
    act[(size_t)row * Icols + n1] = f2bf(g1 * sigmoidf_(g1) * cu1[r]);
  }
}

// ---------------------------------------------------------------- rowwise l2norm (q,k), in-place
__global__ __launch_bounds__(256) void l2norm_kernel(float* __restrict__ x, int rows) {
  int wid = blockIdx.x * 8 + (threadIdx.x >> 5);
  int lane = threadIdx.x & 31;
  if (wid >= rows) return;
  float* p = x + (size_t)(wid / H_) * K_ + (wid % H_) * 64;
  float a = p[lane], b = p[lane + 32];
  float s = a * a + b * b;
  for (int o = 16; o > 0; o >>= 1) s += __shfl_xor(s, o, 32);
  float rs = rsqrtf(s + 1e-6f);
  p[lane] = a * rs; p[lane + 32] = b * rs;
}

// ---------------------------------------------------------------- g finalize: -exp(A_log)*softplus(g+dt_bias)
__global__ void gfin_kernel(float* __restrict__ g, const float* __restrict__ dt_bias,
                            const float* __restrict__ A_log, int n) {
  int i = blockIdx.x * blockDim.x + threadIdx.x;
  int stride = gridDim.x * blockDim.x;
  for (; i < n; i += stride) {
    int col = i & (K_ - 1);
    int h = col >> 6;
    float x = g[i] + dt_bias[col];
    float sp = (x > 20.f) ? x : log1pf(__expf(x));
    g[i] = -__expf(A_log[h]) * sp;
  }
}

// ---------------------------------------------------------------- beta = sigmoid(.) * mask, in-place
__global__ void beta_kernel(float* __restrict__ bet, const int* __restrict__ mask, int n) {
  int i = blockIdx.x * blockDim.x + threadIdx.x;
  int stride = gridDim.x * blockDim.x;
  for (; i < n; i += stride)
    bet[i] = sigmoidf_(bet[i]) * (float)mask[i / H_];
}

// ---------------------------------------------------------------- KDA sequential scan
// One block per (b,h); 64x64 state in registers (16 floats/thread).
__global__ __launch_bounds__(256) void kda_scan_kernel(
    const float* __restrict__ q, const float* __restrict__ k,
    const float* __restrict__ v, const float* __restrict__ g,
    const float* __restrict__ beta, float* __restrict__ o) {
  int bh = blockIdx.x;
  int b = bh / H_, h = bh % H_;
  int tid = threadIdx.x;
  int vcol = tid & 63, kg = tid >> 6;
  float S[16];
#pragma unroll
  for (int i = 0; i < 16; i++) S[i] = 0.f;
  __shared__ float shq[64], shk[64], shg[64], shv[64], sheg[64], shu[64];
  __shared__ float red[4][64];
  __shared__ float shb;
  const float scale = 0.125f;  // DK^-0.5
  for (int t = 0; t < T_; t++) {
    size_t rowK = ((size_t)(b * T_ + t)) * K_ + h * 64;
    size_t rowV = ((size_t)(b * T_ + t)) * V_ + h * 64;
    if (tid < 64) shq[tid] = q[rowK + tid];
    else if (tid < 128) shk[tid - 64] = k[rowK + (tid - 64)];
    else if (tid < 192) shg[tid - 128] = g[rowK + (tid - 128)];
    else shv[tid - 192] = v[rowV + (tid - 192)];
    if (tid == 0) shb = beta[(size_t)(b * T_ + t) * H_ + h];
    __syncthreads();
    if (tid < 64) sheg[tid] = __expf(shg[tid]);
    __syncthreads();
    int kb = kg * 16;
    float part = 0.f;
#pragma unroll
    for (int i = 0; i < 16; i++) { S[i] *= sheg[kb + i]; part += shk[kb + i] * S[i]; }
    red[kg][vcol] = part; __syncthreads();
    if (tid < 64) {
      float verr = shv[tid] - (red[0][tid] + red[1][tid] + red[2][tid] + red[3][tid]);
      shu[tid] = verr * shb;
    }
    __syncthreads();
    float u = shu[vcol];
    float part2 = 0.f;
#pragma unroll
    for (int i = 0; i < 16; i++) { S[i] += shk[kb + i] * u; part2 += shq[kb + i] * S[i]; }
    red[kg][vcol] = part2; __syncthreads();
    if (tid < 64)
      o[rowV + tid] = scale * (red[0][tid] + red[1][tid] + red[2][tid] + red[3][tid]);
    __syncthreads();
  }
}

// ---------------------------------------------------------------- o-norm * o_norm_w * sigmoid(go) -> bf16
__global__ __launch_bounds__(256) void onorm_kernel(
    const float* __restrict__ o, const float* __restrict__ go,
    const float* __restrict__ w, unsigned short* __restrict__ obf, int rows) {
  int wid = blockIdx.x * 8 + (threadIdx.x >> 5);
  int lane = threadIdx.x & 31;
  if (wid >= rows) return;
  size_t base = (size_t)(wid / H_) * V_ + (wid % H_) * 64;
  float a = o[base + lane], b = o[base + lane + 32];
  float s = a * a + b * b;
  for (int off = 16; off > 0; off >>= 1) s += __shfl_xor(s, off, 32);
  float rs = rsqrtf(s * (1.f / 64.f) + 1e-5f);
  float ga = go[base + lane], gb = go[base + lane + 32];
  obf[base + lane]      = f2bf(a * rs * w[lane]      * sigmoidf_(ga));
  obf[base + lane + 32] = f2bf(b * rs * w[lane + 32] * sigmoidf_(gb));
}

// ---------------------------------------------------------------- launch
extern "C" void kernel_launch(void* const* d_in, const int* in_sizes, int n_in,
                              void* d_out, int out_size, void* d_ws, size_t ws_size,
                              hipStream_t stream) {
  (void)in_sizes; (void)n_in; (void)out_size; (void)ws_size;
  const float* hidden   = (const float*)d_in[0];
  const int*   mask     = (const int*)d_in[1];
  const float* ln_w     = (const float*)d_in[2];
  const float* ln_b     = (const float*)d_in[3];
  const float* q_w      = (const float*)d_in[4];
  const float* k_w      = (const float*)d_in[5];
  const float* v_w      = (const float*)d_in[6];
  const float* f_a_w    = (const float*)d_in[7];
  const float* f_b_w    = (const float*)d_in[8];
  const float* dt_bias  = (const float*)d_in[9];
  const float* A_log    = (const float*)d_in[10];
  const float* b_w      = (const float*)d_in[11];
  const float* g_a_w    = (const float*)d_in[12];
  const float* g_b_w    = (const float*)d_in[13];
  const float* o_norm_w = (const float*)d_in[14];
  const float* o_w      = (const float*)d_in[15];
  const float* gu_w     = (const float*)d_in[16];
  const float* down_w   = (const float*)d_in[17];
  float* out = (float*)d_out;

  char* ws = (char*)d_ws;
  size_t off = 0;
  auto alloc = [&](size_t bytes) -> void* {
    void* p = ws + off; off += (bytes + 255) & ~(size_t)255; return p;
  };

  // bf16 weight copies
  unsigned short* wq   = (unsigned short*)alloc((size_t)D_ * K_ * 2);
  unsigned short* wk   = (unsigned short*)alloc((size_t)D_ * K_ * 2);
  unsigned short* wv   = (unsigned short*)alloc((size_t)D_ * V_ * 2);
  unsigned short* wfa  = (unsigned short*)alloc((size_t)D_ * DK_ * 2);
  unsigned short* wfb  = (unsigned short*)alloc((size_t)DK_ * K_ * 2);
  unsigned short* wb   = (unsigned short*)alloc((size_t)D_ * H_ * 2);
  unsigned short* wga  = (unsigned short*)alloc((size_t)D_ * DK_ * 2);
  unsigned short* wgb  = (unsigned short*)alloc((size_t)DK_ * V_ * 2);
  unsigned short* wo   = (unsigned short*)alloc((size_t)V_ * D_ * 2);
  unsigned short* wgu  = (unsigned short*)alloc((size_t)D_ * 2 * I_ * 2);
  unsigned short* wdn  = (unsigned short*)alloc((size_t)I_ * D_ * 2);
  // activations
  unsigned short* xln  = (unsigned short*)alloc((size_t)M_ * D_ * 2);
  float* qf    = (float*)alloc((size_t)M_ * K_ * 4);
  float* kf    = (float*)alloc((size_t)M_ * K_ * 4);
  float* vf    = (float*)alloc((size_t)M_ * V_ * 4);
  float* gf    = (float*)alloc((size_t)M_ * K_ * 4);
  float* faf   = (float*)alloc((size_t)M_ * DK_ * 4);
  unsigned short* fab = (unsigned short*)alloc((size_t)M_ * DK_ * 2);
  float* gaf   = (float*)alloc((size_t)M_ * DK_ * 4);
  unsigned short* gab = (unsigned short*)alloc((size_t)M_ * DK_ * 2);
  float* betap = (float*)alloc((size_t)M_ * H_ * 4);
  float* gof   = (float*)alloc((size_t)M_ * V_ * 4);
  float* of    = (float*)alloc((size_t)M_ * V_ * 4);
  unsigned short* obf = (unsigned short*)alloc((size_t)M_ * V_ * 2);
  float* opf   = (float*)alloc((size_t)M_ * D_ * 4);
  unsigned short* opb = (unsigned short*)alloc((size_t)M_ * D_ * 2);
  unsigned short* act = (unsigned short*)alloc((size_t)M_ * I_ * 2);

  auto conv = [&](const float* s, unsigned short* d, size_t n) {
    int blocks = (int)((n + 2047) / 2048); if (blocks > 4096) blocks = 4096;
    convert_f32_bf16<<<blocks, 256, 0, stream>>>(s, d, (int)n);
  };
  auto gemm = [&](const unsigned short* A, const unsigned short* Bm, float* C,
                  const float* resid, int M, int N, int Kd) {
    dim3 grid((N + 63) / 64, M / 64);
    gemm_bf16_wmma<<<grid, 256, 0, stream>>>(A, Bm, C, resid, M, N, Kd);
  };

  // 1. weights -> bf16
  conv(q_w, wq, (size_t)D_ * K_);
  conv(k_w, wk, (size_t)D_ * K_);
  conv(v_w, wv, (size_t)D_ * V_);
  conv(f_a_w, wfa, (size_t)D_ * DK_);
  conv(f_b_w, wfb, (size_t)DK_ * K_);
  conv(b_w, wb, (size_t)D_ * H_);
  conv(g_a_w, wga, (size_t)D_ * DK_);
  conv(g_b_w, wgb, (size_t)DK_ * V_);
  conv(o_w, wo, (size_t)V_ * D_);
  conv(gu_w, wgu, (size_t)D_ * 2 * I_);
  conv(down_w, wdn, (size_t)I_ * D_);

  // 2. layernorm
  layernorm_kernel<<<M_, 256, 0, stream>>>(hidden, ln_w, ln_b, xln);

  // 3. projections (WMMA GEMMs)
  gemm(xln, wq, qf, nullptr, M_, K_, D_);
  gemm(xln, wk, kf, nullptr, M_, K_, D_);
  gemm(xln, wv, vf, nullptr, M_, V_, D_);
  gemm(xln, wfa, faf, nullptr, M_, DK_, D_);
  conv(faf, fab, (size_t)M_ * DK_);
  gemm(fab, wfb, gf, nullptr, M_, K_, DK_);
  gemm(xln, wb, betap, nullptr, M_, H_, D_);
  gemm(xln, wga, gaf, nullptr, M_, DK_, D_);
  conv(gaf, gab, (size_t)M_ * DK_);
  gemm(gab, wgb, gof, nullptr, M_, V_, DK_);

  // 4. elementwise pre-scan
  l2norm_kernel<<<(M_ * H_) / 8, 256, 0, stream>>>(qf, M_ * H_);
  l2norm_kernel<<<(M_ * H_) / 8, 256, 0, stream>>>(kf, M_ * H_);
  gfin_kernel<<<4096, 256, 0, stream>>>(gf, dt_bias, A_log, M_ * K_);
  beta_kernel<<<256, 256, 0, stream>>>(betap, mask, M_ * H_);

  // 5. sequential KDA scan (one block per (b,h))
  kda_scan_kernel<<<B_ * H_, 256, 0, stream>>>(qf, kf, vf, gf, betap, of);

  // 6. output norm + gating -> bf16
  onorm_kernel<<<(M_ * H_) / 8, 256, 0, stream>>>(of, gof, o_norm_w, obf, M_ * H_);

  // 7. o projection
  gemm(obf, wo, opf, nullptr, M_, D_, V_);
  conv(opf, opb, (size_t)M_ * D_);

  // 8. fused gate_up + SwiGLU
  {
    dim3 grid(I_ / 64, M_ / 64);
    gemm_gateup_wmma<<<grid, 256, 0, stream>>>(opb, wgu, act, I_, D_);
  }

  // 9. down projection + residual -> d_out
  gemm(act, wdn, out, hidden, M_, D_, I_);
}
